// SumOfPDFs_50173807952862
// MI455X (gfx1250) — matrix operand order repaired and checked by reading, not compiled, not run on hardware
//
#include <hip/hip_runtime.h>
#include <hip/hip_bf16.h>
#include <math.h>

// ---------------------------------------------------------------------------
// out[n] = c * exp(-eps*n) * P[min(n,127)]
//   eps = exp(log_epsilon), c = 1 - exp(-eps)
//   P[m] = sum_{j<=m} softmax(logit_pdf_a)[j] * exp(eps*j)
// Head (n<128): prefix sums via lower-triangular WMMA matmul (1 wave).
// Tail (n>=128): store-bandwidth-bound geometric curve, 64MiB -> ~2.9us floor.
// ---------------------------------------------------------------------------

typedef __attribute__((ext_vector_type(16))) _Float16 v16h;
typedef __attribute__((ext_vector_type(8)))  float    v8f;

#define HEAD_K 128

__global__ __launch_bounds__(32) void sumpdf_head(
    const float* __restrict__ log_eps_p,
    const float* __restrict__ logits,
    float* __restrict__ out,
    float* __restrict__ ws)
{
    const int lane = threadIdx.x;              // 0..31, one full wave32
    const float eps = __expf(log_eps_p[0]);
    const float c   = -expm1f(-eps);           // 1 - exp(-eps)

    // ---- softmax over K=128 logits: 4 per lane ------------------------------
    const float4 a4 = *(const float4*)(logits + lane * 4);
    float m = fmaxf(fmaxf(a4.x, a4.y), fmaxf(a4.z, a4.w));
    #pragma unroll
    for (int off = 16; off; off >>= 1) m = fmaxf(m, __shfl_xor(m, off, 32));
    const float e0 = __expf(a4.x - m), e1 = __expf(a4.y - m);
    const float e2 = __expf(a4.z - m), e3 = __expf(a4.w - m);
    float s = e0 + e1 + e2 + e3;
    #pragma unroll
    for (int off = 16; off; off >>= 1) s += __shfl_xor(s, off, 32);
    const float inv = 1.0f / s;

    // ---- v[j] = w[j] * exp(eps*j) -------------------------------------------
    const int j0 = lane * 4;
    const float v0 = e0 * inv * __expf(eps * (float)(j0 + 0));
    const float v1 = e1 * inv * __expf(eps * (float)(j0 + 1));
    const float v2 = e2 * inv * __expf(eps * (float)(j0 + 2));
    const float v3 = e3 * inv * __expf(eps * (float)(j0 + 3));

    // full sum S in f32 (exact path for the tail coefficient)
    float S = v0 + v1 + v2 + v3;
    #pragma unroll
    for (int off = 16; off; off >>= 1) S += __shfl_xor(S, off, 32);

    __shared__ _Float16 vh[HEAD_K];            // pre-converted to f16 for B frags
    vh[j0 + 0] = (_Float16)v0; vh[j0 + 1] = (_Float16)v1;
    vh[j0 + 2] = (_Float16)v2; vh[j0 + 3] = (_Float16)v3;
    __syncthreads();

    // Lane layout used here (contiguous k order, not AMD's table order; A and B agree):
    //  A 16x32: lane L row M=L%16; half h: K = (L<16?0:16)+h
    //  B 32x16: lane L col N=L%16; half h: K = (L<16?0:16)+h
    //  C/D f32: VGPR g -> M = g + (L<16?0:8), N = L%16
    const int nCol   = lane & 15;
    const int mRow   = lane & 15;
    const int hiHalf = (lane >= 16);

    // ---- B fragments for the 4 k-tiles, built once (unconditional loads) ----
    const _Float16 bsel = (nCol == 0) ? (_Float16)1.0f : (_Float16)0.0f;
    const int kBase = hiHalf ? 16 : 0;
    v16h bfr[4];
    #pragma unroll
    for (int kt = 0; kt < 4; ++kt) {
        #pragma unroll
        for (int h = 0; h < 16; ++h) {
            const _Float16 t = vh[32 * kt + kBase + h];  // all lanes load
            bfr[kt][h] = bsel * t;                       // mask in VALU
        }
    }

    // ---- the 3 distinct A fragments of a 128x128 lower-triangular ones ------
    v16h aFull, aEven, aOdd;
    #pragma unroll
    for (int h = 0; h < 16; ++h) {
        const int cc = (hiHalf ? 16 : 0) + h; // local col 0..31
        aFull[h] = (_Float16)1.0f;
        aEven[h] = (cc <= mRow)      ? (_Float16)1.0f : (_Float16)0.0f;
        aOdd[h]  = (cc <= mRow + 16) ? (_Float16)1.0f : (_Float16)0.0f;
    }

    // ---- P = L * v : 20 statically-scheduled WMMAs --------------------------
    #pragma unroll
    for (int i = 0; i < 8; ++i) {              // 8 row tiles of 16
        v8f acc = {};
        const int ktDiag = i >> 1;
        #pragma unroll
        for (int kt = 0; kt < ktDiag; ++kt)    // interior tiles: all-ones A
            acc = __builtin_amdgcn_wmma_f32_16x16x32_f16(
                false, aFull, false, bfr[kt], (short)0, acc, false, false);
        acc = __builtin_amdgcn_wmma_f32_16x16x32_f16(   // diagonal tile
            false, (i & 1) ? aOdd : aEven, false, bfr[ktDiag],
            (short)0, acc, false, false);

        // column N==0 of D holds P[16i + M]; lanes 0 (M=g) and 16 (M=g+8)
        if (nCol == 0) {
            #pragma unroll
            for (int g = 0; g < 8; ++g) {
                const int n = 16 * i + g + (hiHalf ? 8 : 0);
                out[n] = c * __expf(-eps * (float)n) * acc[g];
            }
        }
    }

    if (lane == 0) {
        const float r = __expf(-eps);
        ws[0] = c * S;         // A coefficient for the tail
        ws[1] = eps;
        ws[2] = r;
        ws[3] = r * r;
        ws[4] = r * r * r;
    }
}

__global__ __launch_bounds__(256) void sumpdf_tail(
    const float* __restrict__ ws,
    float* __restrict__ out,
    int amount)
{
    // uniform scalar loads (s_load) of the 5 coefficients
    const float A   = ws[0];
    const float eps = ws[1];
    const float r1 = ws[2], r2 = ws[3], r3 = ws[4];

    const int f  = blockIdx.x * 256 + threadIdx.x;   // float4 index
    const int n0 = HEAD_K + 4 * f;
    if (n0 >= amount) return;

    // one v_exp_f32 per 4 outputs; geometric fill of the float4
    const float e = A * __expf(-eps * (float)n0);
    float4 val;
    val.x = e;
    val.y = e * r1;
    val.z = e * r2;
    val.w = e * r3;
    *(float4*)(out + n0) = val;                      // coalesced global_store_b128
}

extern "C" void kernel_launch(void* const* d_in, const int* in_sizes, int n_in,
                              void* d_out, int out_size, void* d_ws, size_t ws_size,
                              hipStream_t stream) {
    const float* log_eps = (const float*)d_in[0];
    const float* logits  = (const float*)d_in[1];   // K = 128
    float* out = (float*)d_out;
    float* ws  = (float*)d_ws;
    const int amount = out_size;                    // 1<<24

    sumpdf_head<<<1, 32, 0, stream>>>(log_eps, logits, out, ws);

    const int nf4    = (amount - HEAD_K + 3) >> 2;
    const int blocks = (nf4 + 255) / 256;
    sumpdf_tail<<<blocks, 256, 0, stream>>>(ws, out, amount);
}
